// NanoChatAttention_36962488549981
// MI455X (gfx1250) — matrix operand order repaired and checked by reading, not compiled, not run on hardware
//
#include <hip/hip_runtime.h>
#include <hip/hip_bf16.h>

// ---------------------------------------------------------------------------
// Problem constants (NanoChat attention): B=2, S=2048, H=2048, NQ=16, NKV=8, DH=128
// ---------------------------------------------------------------------------
#define BB   2
#define SS   2048
#define HH   2048
#define NQ   16
#define NKV  8
#define DH   128
#define QKVN 4096            // NQ*DH + 2*NKV*DH
#define MTOK 4096            // B*S

typedef __attribute__((ext_vector_type(16))) __bf16 v16bf;
typedef __attribute__((ext_vector_type(8)))  __bf16 v8bf;
typedef __attribute__((ext_vector_type(8)))  float  v8f;

union frag16 { v16bf v; v8bf h[2]; };

__device__ __forceinline__ unsigned short f2bf(float f) {
  unsigned int u = __builtin_bit_cast(unsigned int, f);
  unsigned int r = u + 0x7fffu + ((u >> 16) & 1u);   // round-to-nearest-even
  return (unsigned short)(r >> 16);
}
__device__ __forceinline__ float bf2f(unsigned short h) {
  unsigned int u = ((unsigned int)h) << 16;
  return __builtin_bit_cast(float, u);
}

__device__ __forceinline__ v8f wmma_bf16(v16bf a, v16bf b, v8f c) {
  // D = A(16x32 bf16) * B(32x16 bf16) + C(16x16 f32)
  return __builtin_amdgcn_wmma_f32_16x16x32_bf16(
      /*neg_a=*/false, a, /*neg_b=*/false, b,
      /*c_mod=*/(short)0, c, /*reuse_a=*/false, /*reuse_b=*/false);
}

// CDNA5 async global->LDS DMA copy, 16B per lane, tracked by ASYNCcnt.
// VDST = per-lane LDS byte address, VADDR = 64-bit global address, SADDR=off.
__device__ __forceinline__ void async_copy_b128(unsigned short* lds,
                                                const unsigned short* g) {
  unsigned loff = (unsigned)(size_t)lds;   // generic LDS ptr low 32 = LDS addr
  asm volatile("global_load_async_to_lds_b128 %0, %1, off"
               :: "v"(loff), "v"(g) : "memory");
}

// ---------------------------------------------------------------------------
// f32 -> bf16 conversion
// ---------------------------------------------------------------------------
__global__ void cvt_f32_bf16_kernel(const float* __restrict__ src,
                                    unsigned short* __restrict__ dst, int n) {
  int i = blockIdx.x * 256 + threadIdx.x;
  if (i < n) dst[i] = f2bf(src[i]);
}

// ---------------------------------------------------------------------------
// bf16 GEMM:  C[M,N] = A[M,K] * B[N,K]^T    (A,B bf16 row-major; C f32 or bf16)
// Block: 256 thr (8 waves), tile 128x128, K-step 32, LDS double-buffered via
// global_load_async_to_lds_b128 + s_wait_asynccnt (copy of slab k+1 overlaps
// the 8 WMMAs of slab k).
// Wave (wid&3) -> 32-row strip, (wid>>2) -> 64-col strip: 2x4 WMMA subtiles.
// ---------------------------------------------------------------------------
#define LDS_STRIDE 40   // 32 bf16 + 8 pad (80B rows, 16B aligned)

template <bool OUT_BF16>
__global__ __launch_bounds__(256, 2)
void gemm_bf16_kernel(const unsigned short* __restrict__ A,
                      const unsigned short* __restrict__ Bw,
                      unsigned short* __restrict__ Cb,
                      float* __restrict__ Cf,
                      int M, int N, int K) {
  __shared__ unsigned short sA[2][128 * LDS_STRIDE];
  __shared__ unsigned short sB[2][128 * LDS_STRIDE];

  const int tid  = threadIdx.x;
  const int lane = tid & 31;
  const int wid  = tid >> 5;
  const int bm   = blockIdx.y * 128;
  const int bn   = blockIdx.x * 128;
  const int wm   = (wid & 3) * 32;
  const int wn   = (wid >> 2) * 64;
  const int m16  = lane & 15;
  const int hi   = lane >> 4;

  // per-thread async copy slice: 2 x 16B per matrix per slab (4 async ops)
  auto issue_slab = [&](int buf, int k0) {
#pragma unroll
    for (int it = 0; it < 2; it++) {
      int i = tid + it * 256;            // 0..511
      int r = i >> 2;
      int c = (i & 3) << 3;              // 0,8,16,24 (bf16 units)
      async_copy_b128(&sA[buf][r * LDS_STRIDE + c],
                      &A[(size_t)(bm + r) * K + k0 + c]);
      async_copy_b128(&sB[buf][r * LDS_STRIDE + c],
                      &Bw[(size_t)(bn + r) * K + k0 + c]);
    }
  };

  v8f zero = {0.f, 0.f, 0.f, 0.f, 0.f, 0.f, 0.f, 0.f};
  v8f acc[2][4];
#pragma unroll
  for (int i = 0; i < 2; i++)
#pragma unroll
    for (int j = 0; j < 4; j++) acc[i][j] = zero;

  issue_slab(0, 0);                       // prologue: fill buffer 0

  const int nk = K >> 5;
  for (int ks = 0; ks < nk; ks++) {
    const int cur = ks & 1;
    if (ks + 1 < nk) {
      issue_slab(cur ^ 1, (ks + 1) << 5); // prefetch next slab into other buf
      // 8 outstanding; async ops complete in order -> <=4 means current done
      asm volatile("s_wait_asynccnt 4" ::: "memory");
    } else {
      asm volatile("s_wait_asynccnt 0" ::: "memory");
    }
    __syncthreads();                      // all waves' current slab visible

    frag16 fa[2], fb[4];
#pragma unroll
    for (int i = 0; i < 2; i++) {
      // A frag: row = lane%16, K chunks {hi*8+0..7} and {hi*8+16..23}
      const unsigned short* p =
          &sA[cur][(wm + i * 16 + m16) * LDS_STRIDE + hi * 8];
      fa[i].h[0] = *(const v8bf*)p;
      fa[i].h[1] = *(const v8bf*)(p + 16);
    }
#pragma unroll
    for (int j = 0; j < 4; j++) {
      // B frag: col = lane%16, K chunks {hi*8+0..7} and {hi*8+16..23}
      const unsigned short* p =
          &sB[cur][(wn + j * 16 + m16) * LDS_STRIDE + hi * 8];
      fb[j].h[0] = *(const v8bf*)p;
      fb[j].h[1] = *(const v8bf*)(p + 16);
    }
#pragma unroll
    for (int i = 0; i < 2; i++)
#pragma unroll
      for (int j = 0; j < 4; j++)
        acc[i][j] = wmma_bf16(fa[i].v, fb[j].v, acc[i][j]);
    __syncthreads();                      // done reading before buf reuse
  }

  // D layout: lane -> col = lane%16 (+subtile), rows = hi*8 + e
#pragma unroll
  for (int i = 0; i < 2; i++)
#pragma unroll
    for (int j = 0; j < 4; j++)
#pragma unroll
      for (int e = 0; e < 8; e++) {
        int gr = bm + wm + i * 16 + hi * 8 + e;
        int gc = bn + wn + j * 16 + m16;
        if (OUT_BF16) Cb[(size_t)gr * N + gc] = f2bf(acc[i][j][e]);
        else          Cf[(size_t)gr * N + gc] = acc[i][j][e];
      }
}

// ---------------------------------------------------------------------------
// RoPE (reference convention: sin_e = -sin) + QK L2-norm + layout shuffle.
// One wave per (b,s,slot): slot 0..15 = q head, 16..23 = k head, 24..31 = v.
// Each lane holds d = lane, lane+32, lane+64, lane+96 -> RoPE pair (d, d+64)
// is lane-local (elems 0/2 and 1/3).
// Outputs: qh[b][16][S][DH], kh[b][8][S][DH] (bf16), vt[b][8][DH][S] (bf16).
// ---------------------------------------------------------------------------
__global__ __launch_bounds__(256)
void rope_norm_kernel(const unsigned short* __restrict__ qkv,
                      const float* __restrict__ cosb,
                      const float* __restrict__ sinb,
                      unsigned short* __restrict__ qh,
                      unsigned short* __restrict__ kh,
                      unsigned short* __restrict__ vt) {
  const int row  = blockIdx.x * 8 + (threadIdx.x >> 5);  // 0 .. B*S*32-1
  const int lane = threadIdx.x & 31;
  const int c    = row & 31;
  const int bs   = row >> 5;
  const int s    = bs & (SS - 1);
  const int b    = bs >> 11;

  const unsigned short* src = qkv + (size_t)bs * QKVN + c * DH;
  float x[4];
#pragma unroll
  for (int e = 0; e < 4; e++) x[e] = bf2f(src[lane + e * 32]);

  if (c < 24) {  // q or k: RoPE + l2norm
    const float* cp = cosb + (size_t)bs * DH;
    const float* sp = sinb + (size_t)bs * DH;
    // out[d]    = x1*cos[d]    + x2*sin[d]       (d < 64)
    // out[d+64] = x2*cos[d+64] - x1*sin[d+64]
    {
      float x1 = x[0], x2 = x[2];
      x[0] = x1 * cp[lane] + x2 * sp[lane];
      x[2] = x2 * cp[lane + 64] - x1 * sp[lane + 64];
    }
    {
      float x1 = x[1], x2 = x[3];
      x[1] = x1 * cp[lane + 32] + x2 * sp[lane + 32];
      x[3] = x2 * cp[lane + 96] - x1 * sp[lane + 96];
    }
    float ss = x[0] * x[0] + x[1] * x[1] + x[2] * x[2] + x[3] * x[3];
#pragma unroll
    for (int off = 16; off >= 1; off >>= 1) ss += __shfl_xor(ss, off);
    float scale = rsqrtf(ss * (1.0f / DH) + 1e-6f);
#pragma unroll
    for (int e = 0; e < 4; e++) x[e] *= scale;
  }

  if (c < 16) {            // q head
    unsigned short* dst = qh + ((((size_t)b * NQ + c) * SS + s) * DH);
#pragma unroll
    for (int e = 0; e < 4; e++) dst[lane + e * 32] = f2bf(x[e]);
  } else if (c < 24) {     // k head
    unsigned short* dst = kh + ((((size_t)b * NKV + (c - 16)) * SS + s) * DH);
#pragma unroll
    for (int e = 0; e < 4; e++) dst[lane + e * 32] = f2bf(x[e]);
  } else {                 // v head -> transposed [dh][s]
    unsigned short* dst = vt + (((size_t)b * NKV + (c - 24)) * DH) * SS + s;
#pragma unroll
    for (int e = 0; e < 4; e++) dst[(size_t)(lane + e * 32) * SS] = f2bf(x[e]);
  }
}

// ---------------------------------------------------------------------------
// Causal flash attention with online softmax.
// Block = 128 thr (4 waves); block handles 64 q-rows of one (b,h); each wave
// owns 16 q-rows. K-blocks of 64 columns; uniform trip count across waves.
//  scores: A = Q frag (kept in regs), B = K rows straight from global.
//  PV:     A = P tile via LDS,        B = V^T rows straight from global.
// ---------------------------------------------------------------------------
__global__ __launch_bounds__(128)
void attn_kernel(const unsigned short* __restrict__ qh,
                 const unsigned short* __restrict__ kh,
                 const unsigned short* __restrict__ vt,
                 unsigned short* __restrict__ attn_out) {
  __shared__ float          sS[4][16][68];
  __shared__ unsigned short sP[4][16][72];
  __shared__ float          sScale[4][16];
  __shared__ float          sInvL[4][16];

  const int lane  = threadIdx.x & 31;
  const int wid   = threadIdx.x >> 5;
  const int qtile = blockIdx.x & 31;          // S/64 tiles
  const int hq    = (blockIdx.x >> 5) & 15;   // query head
  const int b     = blockIdx.x >> 9;
  const int kvh   = hq >> 1;                  // GQA: 2 q heads per kv head
  const int qrow0 = qtile * 64 + wid * 16;
  const int m16   = lane & 15;
  const int hi    = lane >> 4;
  const int r     = lane >> 1;                // softmax row owned by lane pair
  const int hf    = lane & 1;
  const int qi    = qrow0 + r;                // absolute query index of row r

  const unsigned short* qb = qh + (((size_t)b * NQ + hq) * SS + qrow0) * DH;
  const unsigned short* kb = kh + ((size_t)b * NKV + kvh) * SS * DH;
  const unsigned short* vb = vt + ((size_t)b * NKV + kvh) * DH * SS;

  // Q fragments for the 4 K-steps over DH (persist in registers)
  frag16 fq[4];
#pragma unroll
  for (int kk = 0; kk < 4; kk++) {
    const unsigned short* p = qb + (size_t)m16 * DH + kk * 32 + hi * 8;
    fq[kk].h[0] = *(const v8bf*)p;
    fq[kk].h[1] = *(const v8bf*)(p + 16);
  }

  v8f zero = {0.f, 0.f, 0.f, 0.f, 0.f, 0.f, 0.f, 0.f};
  v8f o[8];
#pragma unroll
  for (int ns = 0; ns < 8; ns++) o[ns] = zero;
  float m_i = -3.0e38f, l_i = 0.f;

  const int nkb = qtile + 1;                  // causal upper bound (uniform)
  for (int kblk = 0; kblk < nkb; kblk++) {
    const int kcol0 = kblk * 64;

    // ---- scores: 16x64 = 4 subtiles x 4 K-steps --------------------------
    v8f sc[4];
#pragma unroll
    for (int ns = 0; ns < 4; ns++) sc[ns] = zero;
#pragma unroll
    for (int ns = 0; ns < 4; ns++) {
#pragma unroll
      for (int kk = 0; kk < 4; kk++) {
        const unsigned short* p =
            kb + (size_t)(kcol0 + ns * 16 + m16) * DH + kk * 32 + hi * 8;
        frag16 f;
        f.h[0] = *(const v8bf*)p;
        f.h[1] = *(const v8bf*)(p + 16);
        sc[ns] = wmma_bf16(fq[kk].v, f.v, sc[ns]);
      }
    }
#pragma unroll
    for (int ns = 0; ns < 4; ns++)
#pragma unroll
      for (int e = 0; e < 8; e++)
        sS[wid][hi * 8 + e][ns * 16 + m16] =
            sc[ns][e] * 0.08838834764831845f;          // DH^-0.5
    asm volatile("s_wait_dscnt 0" ::: "memory");       // wave-local LDS order

    // ---- online softmax (lane pair per row, 32 cols each) ----------------
    float lm = -3.0e38f;
#pragma unroll
    for (int cc = 0; cc < 32; cc++) {
      int col = kcol0 + hf * 32 + cc;
      float v = (col <= qi) ? sS[wid][r][hf * 32 + cc] : -3.0e38f;
      lm = fmaxf(lm, v);
    }
    lm = fmaxf(lm, __shfl_xor(lm, 1));
    float mnew  = fmaxf(m_i, lm);
    float alpha = __expf(m_i - mnew);
    float ls = 0.f;
#pragma unroll
    for (int cc = 0; cc < 32; cc++) {
      int col = kcol0 + hf * 32 + cc;
      float v  = (col <= qi) ? sS[wid][r][hf * 32 + cc] : -3.0e38f;
      float pr = __expf(v - mnew);
      ls += pr;
      sP[wid][r][hf * 32 + cc] = f2bf(pr);
    }
    ls += __shfl_xor(ls, 1);
    m_i = mnew;
    l_i = l_i * alpha + ls;
    if (hf == 0) sScale[wid][r] = alpha;
    asm volatile("s_wait_dscnt 0" ::: "memory");

    // ---- rescale O and accumulate P @ V ----------------------------------
    float al[8];
#pragma unroll
    for (int e = 0; e < 8; e++) al[e] = sScale[wid][hi * 8 + e];
#pragma unroll
    for (int ns = 0; ns < 8; ns++)
#pragma unroll
      for (int e = 0; e < 8; e++) o[ns][e] *= al[e];

    frag16 fp[2];
#pragma unroll
    for (int kk = 0; kk < 2; kk++) {
      const unsigned short* p = &sP[wid][m16][kk * 32 + hi * 8];
      fp[kk].h[0] = *(const v8bf*)p;
      fp[kk].h[1] = *(const v8bf*)(p + 16);
    }
#pragma unroll
    for (int ns = 0; ns < 8; ns++) {
#pragma unroll
      for (int kk = 0; kk < 2; kk++) {
        const unsigned short* p =
            vb + (size_t)(ns * 16 + m16) * SS + kcol0 + kk * 32 + hi * 8;
        frag16 f;
        f.h[0] = *(const v8bf*)p;
        f.h[1] = *(const v8bf*)(p + 16);
        o[ns] = wmma_bf16(fp[kk].v, f.v, o[ns]);
      }
    }
  }

  // ---- epilogue: O / l, write [b,s,NQ*DH] bf16 ---------------------------
  if (hf == 0) sInvL[wid][r] = 1.0f / l_i;
  asm volatile("s_wait_dscnt 0" ::: "memory");
  float il[8];
#pragma unroll
  for (int e = 0; e < 8; e++) il[e] = sInvL[wid][hi * 8 + e];
#pragma unroll
  for (int ns = 0; ns < 8; ns++)
#pragma unroll
    for (int e = 0; e < 8; e++) {
      int s_idx = qrow0 + hi * 8 + e;
      int col   = hq * DH + ns * 16 + m16;
      attn_out[((size_t)b * SS + s_idx) * (NQ * DH) + col] =
          f2bf(o[ns][e] * il[e]);
    }
}

// ---------------------------------------------------------------------------
// Launch
// ---------------------------------------------------------------------------
extern "C" void kernel_launch(void* const* d_in, const int* in_sizes, int n_in,
                              void* d_out, int out_size, void* d_ws, size_t ws_size,
                              hipStream_t stream) {
  const float* hs   = (const float*)d_in[0];
  const float* cosb = (const float*)d_in[1];
  const float* sinb = (const float*)d_in[2];
  // d_in[3] = attention_mask (causal; applied analytically)
  const float* q_w  = (const float*)d_in[4];
  const float* k_w  = (const float*)d_in[5];
  const float* v_w  = (const float*)d_in[6];
  const float* o_w  = (const float*)d_in[7];
  float* out = (float*)d_out;

  char* ws = (char*)d_ws;
  unsigned short* hs_bf  = (unsigned short*)(ws);                 // 16 MB
  unsigned short* w_qkv  = (unsigned short*)(ws + 16777216);      // 16 MB
  unsigned short* w_o    = (unsigned short*)(ws + 33554432);      //  8 MB
  unsigned short* qkv    = (unsigned short*)(ws + 41943040);      // 32 MB
  unsigned short* qh     = (unsigned short*)(ws + 75497472);      // 16 MB
  unsigned short* kh     = (unsigned short*)(ws + 92274688);      //  8 MB
  unsigned short* vt     = (unsigned short*)(ws + 100663296);     //  8 MB
  unsigned short* attnb  = (unsigned short*)(ws + 109051904);     // 16 MB

  // 1) f32 -> bf16 conversions (hidden states + fused QKV weight + O weight)
  {
    int n = BB * SS * HH;
    cvt_f32_bf16_kernel<<<(n + 255) / 256, 256, 0, stream>>>(hs, hs_bf, n);
    n = HH * HH;
    cvt_f32_bf16_kernel<<<(n + 255) / 256, 256, 0, stream>>>(q_w, w_qkv, n);
    n = NKV * DH * HH;
    cvt_f32_bf16_kernel<<<(n + 255) / 256, 256, 0, stream>>>(
        k_w, w_qkv + (size_t)HH * HH, n);
    cvt_f32_bf16_kernel<<<(n + 255) / 256, 256, 0, stream>>>(
        v_w, w_qkv + (size_t)(HH + NKV * DH) * HH, n);
    n = HH * HH;
    cvt_f32_bf16_kernel<<<(n + 255) / 256, 256, 0, stream>>>(o_w, w_o, n);
  }

  // 2) fused QKV projection: [4096,2048] x [4096,2048]^T -> bf16 [4096,4096]
  gemm_bf16_kernel<true><<<dim3(QKVN / 128, MTOK / 128), 256, 0, stream>>>(
      hs_bf, w_qkv, qkv, nullptr, MTOK, QKVN, HH);

  // 3) RoPE + L2-norm + head-major / V-transpose repack
  rope_norm_kernel<<<(BB * SS * 32) / 8, 256, 0, stream>>>(
      qkv, cosb, sinb, qh, kh, vt);

  // 4) causal flash attention (grid: 32 q-tiles x 16 heads x 2 batches)
  attn_kernel<<<BB * NQ * (SS / 64), 128, 0, stream>>>(qh, kh, vt, attnb);

  // 5) output projection: [4096,2048] x [2048,2048]^T -> f32 d_out
  gemm_bf16_kernel<false><<<dim3(HH / 128, MTOK / 128), 256, 0, stream>>>(
      attnb, w_o, nullptr, out, MTOK, HH, HH);

  (void)in_sizes; (void)n_in; (void)out_size; (void)ws_size;
}
